// SpNCN_Layer_81106162417854
// MI455X (gfx1250) — matrix operand order repaired and checked by reading, not compiled, not run on hardware
//
#include <hip/hip_runtime.h>

// ---------------------------------------------------------------------------
// SpNCN layer on MI455X (gfx1250): per-step GEMM via v_wmma_f32_16x16x32_f16,
// persistent single-wave tiles carrying syn/mem/trace state across T=800.
// Depth-3 software pipeline on A-tile loads; B panel lives in 128 VGPRs.
// ---------------------------------------------------------------------------

typedef __attribute__((ext_vector_type(16))) _Float16 v16h;
typedef __attribute__((ext_vector_type(8)))  float    v8f;

#define T_STEPS 800
#define BATCH   128
#define SDIM    500
#define KPAD    512
#define TBS     ((size_t)T_STEPS * BATCH * SDIM)   // 51,200,000

#define ALPHA 0.975f        // 1 - 0.25/10
#define BETA  0.9875f       // 1 - 0.25/20
#define KAPPA (1.0f - 0.25f/30.0f)
#define THR   0.4f

union V16U { uint4 q[2]; v16h v; };
union V16H { _Float16 h[16]; v16h v; };

// ---- prep: Wh[s][k] = -W_err[s][k] as f16, zero-padded to 512x512 ----------
__global__ void prep_w(const float* __restrict__ W, _Float16* __restrict__ Wh) {
    int idx = blockIdx.x * 256 + threadIdx.x;      // 0 .. 512*512-1
    int s = idx >> 9, k = idx & 511;
    float v = (s < SDIM && k < SDIM) ? -W[s * SDIM + k] : 0.0f;
    Wh[idx] = (_Float16)v;
}

// ---- prep: errh[row][k] = err[row][k] as f16, rows = T*B, k padded to 512 --
__global__ void prep_err(const float* __restrict__ err, _Float16* __restrict__ errh) {
    size_t i = (size_t)blockIdx.x * 256 + threadIdx.x;  // each thread: 4 halfs
    size_t row = i >> 7;                                // 512/4 = 128 chunks/row
    int k4 = (int)(i & 127) * 4;
    const float* src = err + row * SDIM + k4;
    union { _Float16 h[4]; uint2 u; } p;
    if (k4 + 3 < SDIM) {
        float4 f = *(const float4*)src;                 // 16B-aligned (2000%16==0)
        p.h[0] = (_Float16)f.x; p.h[1] = (_Float16)f.y;
        p.h[2] = (_Float16)f.z; p.h[3] = (_Float16)f.w;
    } else {
        #pragma unroll
        for (int j = 0; j < 4; ++j)
            p.h[j] = (k4 + j < SDIM) ? (_Float16)src[j] : (_Float16)0.0f;
    }
    ((uint2*)errh)[i] = p.u;
}

// ---- main persistent kernel: 256 blocks x 32 threads (1 wave / tile) -------
template<bool PRECONV>
__global__ __launch_bounds__(32)
void snn_scan(const float* __restrict__ inp,
              const float* __restrict__ errf,       // fp32 (fallback path)
              const _Float16* __restrict__ errh,    // f16 preconverted
              const _Float16* __restrict__ Wh,
              float* __restrict__ out) {
    const int lane = threadIdx.x;
    const int hi   = lane >> 4;        // 0: lanes 0-15, 1: lanes 16-31
    const int lrow = lane & 15;
    const int b0 = (blockIdx.x & 7) * 16;       // batch tile  (8 tiles)
    const int s0 = (blockIdx.x >> 3) * 16;      // feature tile (32 tiles)
    const int s  = s0 + lrow;                   // this lane's output column
    const bool sok = (s < SDIM);
    const int sc  = sok ? s : (SDIM - 1);       // clamped: branch-free loads

    // --- preload B panel (-W^T columns s0..s0+15) into registers: 128 VGPRs.
    // B 32x16 f16 layout: lane = column N=lrow; K chunks {c..c+7},{c+16..c+23}
    // with c = kb*32 + hi*8 -> uint4 indices kb*4+hi and kb*4+hi+2.
    v16h Breg[16];
    {
        const uint4* wr = (const uint4*)Wh + (size_t)(s0 + lrow) * (KPAD / 8);
        #pragma unroll
        for (int kb = 0; kb < 16; ++kb) {
            V16U t;
            t.q[0] = wr[kb * 4 + hi];
            t.q[1] = wr[kb * 4 + hi + 2];
            Breg[kb] = t.v;
        }
    }

    v8f syn = {}, mem = {}, trace = {};

    // Step-invariant base pointers (advance by one timestep each iteration).
    const float* inp_p = inp + (size_t)(b0 + hi * 8) * SDIM + sc;
    float*       spk_p = out + (size_t)(b0 + hi * 8) * SDIM + s0 + lrow; // guarded
    float*       tr_p  = spk_p + TBS;
    const uint4* ar    = (const uint4*)errh + (size_t)(b0 + lrow) * (KPAD / 8);
    const float* er    = errf + (size_t)(b0 + lrow) * SDIM;

    for (int t = 0; t < T_STEPS; ++t) {
        // inp tile: issued independently of the WMMA chain (acc starts at 0,
        // inp added at the tail), so its latency overlaps all 16 WMMAs.
        float inpv[8];
        #pragma unroll
        for (int j = 0; j < 8; ++j)
            inpv[j] = __builtin_nontemporal_load(&inp_p[j * SDIM]);

        v8f acc = {};

        // K reduction: 16 x wmma_f32_16x16x32_f16, prefetch distance 3.
        if (PRECONV) {
            // A 16x32 layout: lane = row M=lrow, K chunks {c..c+7},{c+16..c+23}
            // with c = kb*32 + hi*8  ->  uint4 indices kb*4+hi and kb*4+hi+2.
            V16U abuf[4];
            #pragma unroll
            for (int p = 0; p < 3; ++p) {
                abuf[p].q[0] = ar[p * 4 + hi];
                abuf[p].q[1] = ar[p * 4 + hi + 2];
            }
            #pragma unroll
            for (int kb = 0; kb < 16; ++kb) {
                if (kb + 3 < 16) {
                    abuf[(kb + 3) & 3].q[0] = ar[(kb + 3) * 4 + hi];
                    abuf[(kb + 3) & 3].q[1] = ar[(kb + 3) * 4 + hi + 2];
                }
                acc = __builtin_amdgcn_wmma_f32_16x16x32_f16(
                        false, abuf[kb & 3].v, false, Breg[kb],
                        (short)0, acc, false, false);
            }
        } else {
            #pragma unroll
            for (int kb = 0; kb < 16; ++kb) {
                const int k1 = kb * 32 + hi * 8;
                V16H a;
                if (kb < 15) {
                    #pragma unroll
                    for (int i = 0; i < 8; ++i) a.h[i]     = (_Float16)er[k1 + i];
                    #pragma unroll
                    for (int i = 0; i < 8; ++i) a.h[8 + i] = (_Float16)er[k1 + 16 + i];
                } else {
                    #pragma unroll
                    for (int i = 0; i < 8; ++i) {
                        int ka = k1 + i, kb2 = k1 + 16 + i;
                        a.h[i]     = (ka  < SDIM) ? (_Float16)er[ka]  : (_Float16)0.0f;
                        a.h[8 + i] = (kb2 < SDIM) ? (_Float16)er[kb2] : (_Float16)0.0f;
                    }
                }
                acc = __builtin_amdgcn_wmma_f32_16x16x32_f16(
                        false, a.v, false, Breg[kb], (short)0, acc, false, false);
            }
        }

        // Neuron state update; outputs streamed with non-temporal stores under
        // a single per-step exec-mask region (lanes with s >= 500 masked off).
        float spkv[8], trv[8];
        #pragma unroll
        for (int j = 0; j < 8; ++j) {
            float total = inpv[j] + acc[j];             // inp - err @ W^T
            float sy = ALPHA * syn[j] + total;
            float keep = (mem[j] > THR) ? 0.0f : 1.0f;  // 1 - reset(mem_prev)
            float mm = (BETA * mem[j] + sy) * keep;
            float spk = (mm > THR) ? 1.0f : 0.0f;
            float tr = KAPPA * trace[j] + spk;
            syn[j] = sy; mem[j] = mm; trace[j] = tr;
            spkv[j] = spk; trv[j] = tr;
        }
        if (sok) {
            #pragma unroll
            for (int j = 0; j < 8; ++j) {
                __builtin_nontemporal_store(spkv[j], &spk_p[j * SDIM]);
                __builtin_nontemporal_store(trv[j],  &tr_p[j * SDIM]);
            }
        }

        inp_p += (size_t)BATCH * SDIM;
        spk_p += (size_t)BATCH * SDIM;
        tr_p  += (size_t)BATCH * SDIM;
        ar    += (size_t)BATCH * (KPAD / 8);
        er    += (size_t)BATCH * SDIM;
    }
}

extern "C" void kernel_launch(void* const* d_in, const int* in_sizes, int n_in,
                              void* d_out, int out_size, void* d_ws, size_t ws_size,
                              hipStream_t stream) {
    const float* inp = (const float*)d_in[0];
    const float* err = (const float*)d_in[1];
    const float* W   = (const float*)d_in[2];
    float* out = (float*)d_out;

    const size_t w_bytes   = (size_t)KPAD * KPAD * sizeof(_Float16);            // 512 KB
    const size_t err_bytes = (size_t)T_STEPS * BATCH * KPAD * sizeof(_Float16); // ~105 MB
    _Float16* Wh   = (_Float16*)d_ws;
    _Float16* errh = (_Float16*)((char*)d_ws + w_bytes);

    prep_w<<<(KPAD * KPAD) / 256, 256, 0, stream>>>(W, Wh);

    const bool preconv = (ws_size >= w_bytes + err_bytes);
    if (preconv) {
        const size_t n4 = (size_t)T_STEPS * BATCH * KPAD / 4;   // 13,107,200
        prep_err<<<(unsigned)(n4 / 256), 256, 0, stream>>>(err, errh);
        snn_scan<true><<<256, 32, 0, stream>>>(inp, err, errh, Wh, out);
    } else {
        snn_scan<false><<<256, 32, 0, stream>>>(inp, err, errh, Wh, out);
    }
}
